// FuncConv_52209622450432
// MI455X (gfx1250) — matrix operand order repaired and checked by reading, not compiled, not run on hardware
//
#include <hip/hip_runtime.h>
#include <hip/hip_bf16.h>

// ---------------------------------------------------------------------------
// FuncConv for MI455X (gfx1250): WMMA f16 GEMMs + f32 atomic scatter-mean.
//
// Roofline: ~23 G-MACs of GEMM (nothing for WMMA pipes) vs ~1.6 GB of edge
// gather/scatter traffic -> memory bound on the edge phase. Precomputing
// mlp3_inv per-node (instead of per-edge) cuts GEMM work 8x. All dense math
// runs through v_wmma_f32_16x16x32_f16 (f16 A/B, f32 accum).
// ---------------------------------------------------------------------------

typedef _Float16 v8h  __attribute__((ext_vector_type(8)));
typedef _Float16 v16h __attribute__((ext_vector_type(16)));
typedef float    v8f  __attribute__((ext_vector_type(8)));

#define NNODES 50000
#define EDGES  800000
#define HDIM   256
#define HHALF  128
#define ODIM   256

// -------------------------------- utilities --------------------------------

__global__ void zero_f32_kernel(float* __restrict__ p, long n) {
  long i = (long)blockIdx.x * blockDim.x + threadIdx.x;
  long stride = (long)gridDim.x * blockDim.x;
  for (; i < n; i += stride) p[i] = 0.0f;
}

__global__ void cast_f16_kernel(const float* __restrict__ x,
                                _Float16* __restrict__ y, long n) {
  long i = (long)blockIdx.x * blockDim.x + threadIdx.x;
  long stride = (long)gridDim.x * blockDim.x;
  for (; i < n; i += stride) y[i] = (_Float16)x[i];
}

// W is [K x Nc] row-major f32; produce Wt [Nc x K] f16 (column-major original)
// so WMMA B-fragments load as contiguous K runs per output column.
__global__ void transpose_f16_kernel(const float* __restrict__ W,
                                     _Float16* __restrict__ Wt,
                                     int K, int Nc) {
  long n = (long)K * Nc;
  long i = (long)blockIdx.x * blockDim.x + threadIdx.x;
  long stride = (long)gridDim.x * blockDim.x;
  for (; i < n; i += stride) {
    int k = (int)(i / Nc);
    int c = (int)(i % Nc);
    Wt[(long)c * K + k] = (_Float16)W[i];
  }
}

// ------------------------------ WMMA GEMM ----------------------------------
// One wave per 16x16 output tile of  out = act(A[M x K] @ W[K x Nc] + bias).
// A: row-major f16. Bt: [Nc x K] f16 (W transposed). M % 16 == 0, K % 32 == 0.
//
// A fragment (16x32 f16, ISA 7.12.2): lane l -> row (l&15); halves 0..7 hold
// K = (l>=16?8:0)+0..7, halves 8..15 hold K = 16+(l>=16?8:0)+0..7.
// B fragment (32x16 f16): lane l -> col (l&15); same K order as A
// (halves 0..7: (l>=16?8:0)+0..7, halves 8..15: 16+(l>=16?8:0)+0..7).
// C/D (16x16 f32): lane l -> col (l&15); acc[j] -> row (l>=16?8:0)+j.
__global__ __launch_bounds__(32)
void gemm_wmma_kernel(const _Float16* __restrict__ A,
                      const _Float16* __restrict__ Bt,
                      const float* __restrict__ bias,
                      _Float16* __restrict__ out16,   // one of out16/out32
                      float* __restrict__ out32,      //   is null
                      int K, int Nc, int act)         // 0 none, 1 leaky, 2 relu
{
  const int lane = threadIdx.x & 31;
  const int row0 = blockIdx.x * 16;
  const int col0 = blockIdx.y * 16;
  const int lo   = lane & 15;
  const int hi8  = (lane >> 4) * 8;    // A per-lane-half K offset
  const int hi16 = (lane >> 4) * 16;   // unused (B also uses hi8)

  const _Float16* Ap = A  + (long)(row0 + lo) * K;
  const _Float16* Bp = Bt + (long)(col0 + lo) * K;

  v8f acc = {};
#pragma unroll 4
  for (int ks = 0; ks < K; ks += 32) {
    v8h a0 = *(const v8h*)(Ap + ks + hi8);        // K = ks+hi8 .. +7
    v8h a1 = *(const v8h*)(Ap + ks + 16 + hi8);   // K = ks+16+hi8 .. +7
    v16h a = __builtin_shufflevector(a0, a1,
               0, 1, 2, 3, 4, 5, 6, 7, 8, 9, 10, 11, 12, 13, 14, 15);
    v8h b0 = *(const v8h*)(Bp + ks + hi8);        // K = ks+hi8 .. +7
    v8h b1 = *(const v8h*)(Bp + ks + 16 + hi8);   // K = ks+16+hi8 .. +7
    v16h b = __builtin_shufflevector(b0, b1,
               0, 1, 2, 3, 4, 5, 6, 7, 8, 9, 10, 11, 12, 13, 14, 15);
    acc = __builtin_amdgcn_wmma_f32_16x16x32_f16(
        /*neg_a=*/false, a, /*neg_b=*/false, b,
        /*c_mod=*/(short)0, acc, /*reuse_a=*/false, /*reuse_b=*/false);
  }

  const float bn = bias[col0 + lo];
#pragma unroll
  for (int j = 0; j < 8; ++j) {
    float v = acc[j] + bn;
    if (act == 1)      v = (v >= 0.0f) ? v : 0.01f * v;   // LeakyReLU(0.01)
    else if (act == 2) v = (v >= 0.0f) ? v : 0.0f;        // ReLU
    const long idx = (long)(row0 + hi8 + j) * Nc + (col0 + lo);
    if (out16) out16[idx] = (_Float16)v;
    else       out32[idx] = v;
  }
}

// --------------------------- edge scatter (mean) ---------------------------
// 8 threads per edge, 32 features each. msg = (r==1 ? invh : h)[src]; atomic
// f32 adds into sums[dst], plus one count per edge.
__global__ void edge_scatter_kernel(const _Float16* __restrict__ h16,
                                    const _Float16* __restrict__ invh16,
                                    const int* __restrict__ src,
                                    const int* __restrict__ dst,
                                    const int* __restrict__ r,
                                    float* __restrict__ sums,
                                    float* __restrict__ cnt) {
  long t = (long)blockIdx.x * blockDim.x + threadIdx.x;
  const long total = (long)EDGES * 8;
  if (t >= total) return;
  const int e = (int)(t >> 3);
  const int c = (int)(t & 7) * 32;
  const int s = src[e];
  const int d = dst[e];
  const _Float16* sp = ((r[e] == 1) ? invh16 : h16) + (long)s * HDIM + c;
  float* dp = sums + (long)d * HDIM + c;
#pragma unroll
  for (int i = 0; i < 32; ++i)
    atomicAdd(dp + i, (float)sp[i]);
  if ((t & 7) == 0)
    atomicAdd(cnt + d, 1.0f);
}

__global__ void neigh_kernel(const float* __restrict__ sums,
                             const float* __restrict__ cnt,
                             _Float16* __restrict__ neigh16) {
  long n = (long)NNODES * HDIM;
  long i = (long)blockIdx.x * blockDim.x + threadIdx.x;
  long stride = (long)gridDim.x * blockDim.x;
  for (; i < n; i += stride) {
    float c = cnt[i >> 8];          // HDIM == 256
    c = (c < 1.0f) ? 1.0f : c;
    neigh16[i] = (_Float16)(sums[i] / c);
  }
}

__global__ void select_kernel(_Float16* __restrict__ res16,
                              const _Float16* __restrict__ resinv16,
                              const int* __restrict__ inv) {
  long n = (long)NNODES * HDIM;
  long i = (long)blockIdx.x * blockDim.x + threadIdx.x;
  long stride = (long)gridDim.x * blockDim.x;
  for (; i < n; i += stride)
    if (inv[i >> 8] == 1) res16[i] = resinv16[i];
}

// ------------------------------- launcher ----------------------------------

extern "C" void kernel_launch(void* const* d_in, const int* in_sizes, int n_in,
                              void* d_out, int out_size, void* d_ws, size_t ws_size,
                              hipStream_t stream) {
  (void)in_sizes; (void)n_in; (void)out_size; (void)ws_size;

  const float* h       = (const float*)d_in[0];
  const int*   src     = (const int*)d_in[1];
  const int*   dst     = (const int*)d_in[2];
  const int*   r       = (const int*)d_in[3];
  const int*   inv     = (const int*)d_in[4];
  const float* inv_W1  = (const float*)d_in[5];
  const float* inv_b1  = (const float*)d_in[6];
  const float* inv_W2  = (const float*)d_in[7];
  const float* inv_b2  = (const float*)d_in[8];
  const float* inv_W3  = (const float*)d_in[9];
  const float* inv_b3  = (const float*)d_in[10];
  const float* and_W1  = (const float*)d_in[11];
  const float* and_b1  = (const float*)d_in[12];
  const float* and_W2  = (const float*)d_in[13];
  const float* and_b2  = (const float*)d_in[14];
  const float* and_W3  = (const float*)d_in[15];
  const float* and_b3  = (const float*)d_in[16];
  const float* proj_W1 = (const float*)d_in[17];
  const float* proj_b1 = (const float*)d_in[18];
  const float* proj_W2 = (const float*)d_in[19];
  const float* proj_b2 = (const float*)d_in[20];
  float* out = (float*)d_out;

  // ---- workspace layout (256B aligned slices) ----
  char* ws = (char*)d_ws;
  size_t off = 0;
  auto take = [&](size_t bytes) -> char* {
    char* p = ws + off;
    off += (bytes + 255) & ~(size_t)255;
    return p;
  };
  _Float16* w_inv1 = (_Float16*)take((size_t)HHALF * HDIM  * 2); // [128 x 256]
  _Float16* w_inv2 = (_Float16*)take((size_t)HHALF * HHALF * 2); // [128 x 128]
  _Float16* w_inv3 = (_Float16*)take((size_t)HDIM  * HHALF * 2); // [256 x 128]
  _Float16* w_and1 = (_Float16*)take((size_t)HHALF * HDIM  * 2);
  _Float16* w_and2 = (_Float16*)take((size_t)HHALF * HHALF * 2);
  _Float16* w_and3 = (_Float16*)take((size_t)HDIM  * HHALF * 2);
  _Float16* w_prj1 = (_Float16*)take((size_t)HDIM  * HDIM  * 2); // [256 x 256]
  _Float16* w_prj2 = (_Float16*)take((size_t)ODIM  * HDIM  * 2);
  _Float16* h16     = (_Float16*)take((size_t)NNODES * HDIM  * 2);
  _Float16* invh16  = (_Float16*)take((size_t)NNODES * HDIM  * 2);
  _Float16* t1      = (_Float16*)take((size_t)NNODES * HHALF * 2);
  _Float16* t2      = (_Float16*)take((size_t)NNODES * HHALF * 2);
  float*    sums    = (float*)   take((size_t)NNODES * HDIM  * 4);
  float*    cnt     = (float*)   take((size_t)NNODES * 4);
  _Float16* neigh16 = (_Float16*)take((size_t)NNODES * HDIM  * 2);
  _Float16* res16   = (_Float16*)take((size_t)NNODES * HDIM  * 2);
  _Float16* rinv16  = (_Float16*)take((size_t)NNODES * HDIM  * 2);
  _Float16* p1      = (_Float16*)take((size_t)NNODES * HDIM  * 2);

  const long NH = (long)NNODES * HDIM;
  const int EB = 256;
  const int gNH = 4096;  // grid-stride elementwise grids

  // ---- zero accumulators (workspace is poisoned; must re-zero every call) ----
  zero_f32_kernel<<<gNH, EB, 0, stream>>>(sums, NH);
  zero_f32_kernel<<<64, EB, 0, stream>>>(cnt, (long)NNODES);

  // ---- weight conversion (f32 -> f16, transposed to [out x in]) ----
  transpose_f16_kernel<<<256, EB, 0, stream>>>(inv_W1,  w_inv1, HDIM,  HHALF);
  transpose_f16_kernel<<<128, EB, 0, stream>>>(inv_W2,  w_inv2, HHALF, HHALF);
  transpose_f16_kernel<<<256, EB, 0, stream>>>(inv_W3,  w_inv3, HHALF, HDIM);
  transpose_f16_kernel<<<256, EB, 0, stream>>>(and_W1,  w_and1, HDIM,  HHALF);
  transpose_f16_kernel<<<128, EB, 0, stream>>>(and_W2,  w_and2, HHALF, HHALF);
  transpose_f16_kernel<<<256, EB, 0, stream>>>(and_W3,  w_and3, HHALF, HDIM);
  transpose_f16_kernel<<<512, EB, 0, stream>>>(proj_W1, w_prj1, HDIM,  HDIM);
  transpose_f16_kernel<<<512, EB, 0, stream>>>(proj_W2, w_prj2, HDIM,  ODIM);

  cast_f16_kernel<<<gNH, EB, 0, stream>>>(h, h16, NH);

  const int RT = NNODES / 16;          // 3125 row tiles (exact)
  dim3 gH (RT, HDIM  / 16);            // Nc = 256 -> 16 col tiles
  dim3 gHh(RT, HHALF / 16);            // Nc = 128 ->  8 col tiles

  // ---- invh = mlp3_inv(h) : hoisted out of the edge loop ----
  gemm_wmma_kernel<<<gHh, 32, 0, stream>>>(h16, w_inv1, inv_b1, t1, nullptr, HDIM,  HHALF, 1);
  gemm_wmma_kernel<<<gHh, 32, 0, stream>>>(t1,  w_inv2, inv_b2, t2, nullptr, HHALF, HHALF, 1);
  gemm_wmma_kernel<<<gH,  32, 0, stream>>>(t2,  w_inv3, inv_b3, invh16, nullptr, HHALF, HDIM, 0);

  // ---- edge gather + atomic scatter-add (mean aggregation) ----
  {
    long work = (long)EDGES * 8;
    int blocks = (int)((work + EB - 1) / EB);
    edge_scatter_kernel<<<blocks, EB, 0, stream>>>(h16, invh16, src, dst, r, sums, cnt);
  }
  neigh_kernel<<<gNH, EB, 0, stream>>>(sums, cnt, neigh16);

  // ---- res = mlp3_and(neigh) ----
  gemm_wmma_kernel<<<gHh, 32, 0, stream>>>(neigh16, w_and1, and_b1, t1, nullptr, HDIM,  HHALF, 1);
  gemm_wmma_kernel<<<gHh, 32, 0, stream>>>(t1,      w_and2, and_b2, t2, nullptr, HHALF, HHALF, 1);
  gemm_wmma_kernel<<<gH,  32, 0, stream>>>(t2,      w_and3, and_b3, res16, nullptr, HHALF, HDIM, 0);

  // ---- res = where(inv==1, mlp3_inv(res), res) ----
  gemm_wmma_kernel<<<gHh, 32, 0, stream>>>(res16, w_inv1, inv_b1, t1, nullptr, HDIM,  HHALF, 1);
  gemm_wmma_kernel<<<gHh, 32, 0, stream>>>(t1,    w_inv2, inv_b2, t2, nullptr, HHALF, HHALF, 1);
  gemm_wmma_kernel<<<gH,  32, 0, stream>>>(t2,    w_inv3, inv_b3, rinv16, nullptr, HHALF, HDIM, 0);
  select_kernel<<<gNH, EB, 0, stream>>>(res16, rinv16, inv);

  // ---- out = relu(res @ proj_W1 + b1) @ proj_W2 + b2  (f32 out) ----
  gemm_wmma_kernel<<<gH, 32, 0, stream>>>(res16, w_prj1, proj_b1, p1, nullptr, HDIM, HDIM, 2);
  gemm_wmma_kernel<<<gH, 32, 0, stream>>>(p1,    w_prj2, proj_b2, nullptr, out, HDIM, ODIM, 0);
}
